// Encoder_GLSearch_67912022884656
// MI455X (gfx1250) — compile-verified, matrix operand order checked by reading, not run on hardware
//
#include <hip/hip_runtime.h>
#include <hip/hip_bf16.h>

#define N_NODES 30000
#define E_EDGES 480000
#define EN_TOT  (E_EDGES + N_NODES)   // edges + self loops
#define DIM     128
#define NLAYERS 3
#define NEG_SLOPE 0.2f
#define NWEIGHTS 4                    // W_mlp + 3 layer weights
#define WPK_ELEMS_PER_W (8 * 4 * 32 * 16)   // ct * kstep * lane * halves = 16384

typedef __attribute__((ext_vector_type(16))) __bf16 v16bf;
typedef __attribute__((ext_vector_type(4)))  __bf16 v4bf;
typedef __attribute__((ext_vector_type(8)))  float  v8f;

// ---- monotonic float<->uint mapping so atomicMax(uint) == float max ----
__device__ __forceinline__ unsigned f2ord(float x) {
    unsigned b = __float_as_uint(x);
    return (b & 0x80000000u) ? ~b : (b | 0x80000000u);
}
__device__ __forceinline__ float ord2f(unsigned u) {
    return __uint_as_float((u & 0x80000000u) ? (u ^ 0x80000000u) : ~u);
}

// ---------------------------------------------------------------------------
// Pre-swizzle all weight matrices (f32 K-major [128x128]) into bf16 laid out
// exactly as the WMMA B operand expects: [w][ct][s][lane][16 halves].
// For lane l (r=l&15, hi=l>>4), half h: K = s*32 + (h<8 ? 8*hi+h : 16+8*hi+h-8),
// value = W[K][ct*16 + r]. One wave per (w, ct, s).
// ---------------------------------------------------------------------------
__global__ void __launch_bounds__(256)
pack_weights_kernel(const float* __restrict__ W_mlp, const float* __restrict__ Ws,
                    __bf16* __restrict__ Wpk)
{
    const int gw   = (blockIdx.x * blockDim.x + threadIdx.x) >> 5; // 0..127
    const int lane = threadIdx.x & 31;
    if (gw >= NWEIGHTS * 32) return;
    const int w  = gw >> 5;          // weight index 0..3
    const int ct = (gw & 31) >> 2;   // column tile 0..7
    const int s  = gw & 3;           // k-step 0..3
    const float* W = (w == 0) ? W_mlp : (Ws + (size_t)(w - 1) * DIM * DIM);
    const int r  = lane & 15;
    const int hi = lane >> 4;
    const int col = ct * 16 + r;
    const int kb  = s * 32;
    __bf16* out = Wpk + ((size_t)gw * 32 + lane) * 16;
#pragma unroll
    for (int h = 0; h < 16; ++h) {
        const int K = kb + ((h < 8) ? (8 * hi + h) : (16 + 8 * hi + (h - 8)));
        out[h] = (__bf16)W[(size_t)K * DIM + col];
    }
}

// ---------------------------------------------------------------------------
// WMMA GEMM: out[rows x 128] = X[rows x 128] * W (pre-swizzled bf16 B operand).
// A converted f32->bf16 on load, f32 accumulate. One wave per 16x16 tile,
// 8 waves/block = the 8 column tiles of one 16-row strip. K = 4 x 32.
// BF16_OUT=true  -> store bf16 H (no bias);  false -> store f32 + bias.
// ---------------------------------------------------------------------------
template <bool BF16_OUT>
__global__ void __launch_bounds__(256)
gemm128_wmma(const float* __restrict__ X, const __bf16* __restrict__ Wpk,
             const float* __restrict__ bias, void* __restrict__ out, int rows)
{
    if (blockIdx.x * 16 >= (unsigned)rows) return;   // uniform guard: EXEC stays all-ones
    const int lane = threadIdx.x & 31;
    const int ct   = threadIdx.x >> 5;               // column tile 0..7
    const int r    = lane & 15;                      // A-row / D-col within tile
    const int hi   = lane >> 4;                      // K-half selector
    const int row  = blockIdx.x * 16 + r;
    const int col0 = ct * 16;

    v8f acc = {};
#pragma unroll
    for (int s = 0; s < 4; ++s) {
        const int kb = s * 32;
        // A 16x32 bf16: halves 0..7 -> K = kb+8*hi+j ; halves 8..15 -> K = kb+16+8*hi+j
        v16bf a;
        const float* ap = X + (size_t)row * DIM + kb + 8 * hi;
#pragma unroll
        for (int j = 0; j < 8; ++j) {
            a[j]     = (__bf16)ap[j];
            a[8 + j] = (__bf16)ap[16 + j];
        }
        // B: one contiguous 32-byte pre-swizzled load per lane
        const v16bf b = *(const v16bf*)(Wpk + ((size_t)(ct * 4 + s) * 32 + lane) * 16);
        acc = __builtin_amdgcn_wmma_f32_16x16x32_bf16(
            /*neg_a=*/false, a, /*neg_b=*/false, b,
            /*c_mod=*/(short)0, acc, /*reuse_a=*/false, /*reuse_b=*/false);
    }
    // D: VGPR v -> row v + 8*hi, column = col0 + (lane&15)
    if (BF16_OUT) {
        __bf16* H = (__bf16*)out;
#pragma unroll
        for (int v = 0; v < 8; ++v) {
            const int m = blockIdx.x * 16 + v + 8 * hi;
            H[(size_t)m * DIM + col0 + r] = (__bf16)acc[v];
        }
    } else {
        float* H = (float*)out;
        const float bv = bias[col0 + r];
#pragma unroll
        for (int v = 0; v < 8; ++v) {
            const int m = blockIdx.x * 16 + v + 8 * hi;
            H[(size_t)m * DIM + col0 + r] = acc[v] + bv;
        }
    }
}

// ---------------------------------------------------------------------------
// Per-node: alpha_s = H.asrc, alpha_d = H.adst (one wave per row, 4 bf16/lane)
// Also zero-initializes m (ordered), z, and the 128-wide accumulator row.
// ---------------------------------------------------------------------------
__global__ void __launch_bounds__(256)
alpha_init_kernel(const __bf16* __restrict__ Hb,
                  const float* __restrict__ asrc, const float* __restrict__ adst,
                  float* __restrict__ as_, float* __restrict__ ad_,
                  unsigned* __restrict__ m_u, float* __restrict__ z,
                  float* __restrict__ accum)
{
    const int wave = blockIdx.x * (blockDim.x >> 5) + (threadIdx.x >> 5);
    const int lane = threadIdx.x & 31;
    if (wave >= N_NODES) return;

    const v4bf   h4 = ((const v4bf*)(Hb + (size_t)wave * DIM))[lane];
    const float4 s4 = ((const float4*)asrc)[lane];
    const float4 d4 = ((const float4*)adst)[lane];
    const float hx = (float)h4[0], hy = (float)h4[1], hz = (float)h4[2], hw = (float)h4[3];
    float ps = hx * s4.x + hy * s4.y + hz * s4.z + hw * s4.w;
    float pd = hx * d4.x + hy * d4.y + hz * d4.z + hw * d4.w;
#pragma unroll
    for (int off = 16; off > 0; off >>= 1) {
        ps += __shfl_down(ps, off, 32);
        pd += __shfl_down(pd, off, 32);
    }
    if (lane == 0) {
        as_[wave] = ps;
        ad_[wave] = pd;
        m_u[wave] = 0u;       // below any ordered-mapped real value
        z[wave]   = 0.0f;
    }
    ((float4*)(accum + (size_t)wave * DIM))[lane] = make_float4(0.f, 0.f, 0.f, 0.f);
}

// ---------------------------------------------------------------------------
// Edge pass 1: segment max of leaky_relu(as[src]+ad[dst]) via ordered-uint max
// ---------------------------------------------------------------------------
__global__ void __launch_bounds__(256)
edge_max_kernel(const int* __restrict__ esrc, const int* __restrict__ edst,
                const float* __restrict__ as_, const float* __restrict__ ad_,
                unsigned* __restrict__ m_u)
{
    const int e = blockIdx.x * blockDim.x + threadIdx.x;
    if (e >= EN_TOT) return;
    int s, d;
    if (e < E_EDGES) { s = esrc[e]; d = edst[e]; } else { s = d = e - E_EDGES; }
    float v = as_[s] + ad_[d];
    v = (v > 0.0f) ? v : NEG_SLOPE * v;
    atomicMax(&m_u[d], f2ord(v));
}

// ---------------------------------------------------------------------------
// Edge pass 2: z[dst] += exp(e - m[dst])
// ---------------------------------------------------------------------------
__global__ void __launch_bounds__(256)
edge_sum_kernel(const int* __restrict__ esrc, const int* __restrict__ edst,
                const float* __restrict__ as_, const float* __restrict__ ad_,
                const unsigned* __restrict__ m_u, float* __restrict__ z)
{
    const int e = blockIdx.x * blockDim.x + threadIdx.x;
    if (e >= EN_TOT) return;
    int s, d;
    if (e < E_EDGES) { s = esrc[e]; d = edst[e]; } else { s = d = e - E_EDGES; }
    float v = as_[s] + ad_[d];
    v = (v > 0.0f) ? v : NEG_SLOPE * v;
    atomicAdd(&z[d], __expf(v - ord2f(m_u[d])));
}

// ---------------------------------------------------------------------------
// Edge pass 3: accum[dst] += (exp(e-m)/z) * H[src]; one wave per edge,
// 32 lanes x 4 bf16 = 128-wide coalesced gather, f32 atomic accumulate.
// ---------------------------------------------------------------------------
__global__ void __launch_bounds__(256)
edge_aggr_kernel(const int* __restrict__ esrc, const int* __restrict__ edst,
                 const float* __restrict__ as_, const float* __restrict__ ad_,
                 const unsigned* __restrict__ m_u, const float* __restrict__ z,
                 const __bf16* __restrict__ Hb, float* __restrict__ accum)
{
    const int wave = blockIdx.x * (blockDim.x >> 5) + (threadIdx.x >> 5);
    const int lane = threadIdx.x & 31;
    if (wave >= EN_TOT) return;
    int s, d;
    if (wave < E_EDGES) { s = esrc[wave]; d = edst[wave]; } else { s = d = wave - E_EDGES; }
    float v = as_[s] + ad_[d];
    v = (v > 0.0f) ? v : NEG_SLOPE * v;
    const float coef = __expf(v - ord2f(m_u[d])) / z[d];

    const v4bf h4 = ((const v4bf*)(Hb + (size_t)s * DIM))[lane];
    float* ap = accum + (size_t)d * DIM + lane * 4;
    atomicAdd(ap + 0, coef * (float)h4[0]);
    atomicAdd(ap + 1, coef * (float)h4[1]);
    atomicAdd(ap + 2, coef * (float)h4[2]);
    atomicAdd(ap + 3, coef * (float)h4[3]);
}

// ---------------------------------------------------------------------------
// Finalize: X = elu(accum + bias)
// ---------------------------------------------------------------------------
__global__ void __launch_bounds__(256)
finalize_kernel(const float* __restrict__ accum, const float* __restrict__ bias,
                float* __restrict__ X)
{
    const int i = blockIdx.x * blockDim.x + threadIdx.x;
    if (i >= N_NODES * DIM) return;
    const float v = accum[i] + bias[i & (DIM - 1)];
    X[i] = (v > 0.0f) ? v : (__expf(v) - 1.0f);
}

// ---------------------------------------------------------------------------
extern "C" void kernel_launch(void* const* d_in, const int* in_sizes, int n_in,
                              void* d_out, int out_size, void* d_ws, size_t ws_size,
                              hipStream_t stream)
{
    (void)in_sizes; (void)n_in; (void)out_size; (void)ws_size;

    const float* xq    = (const float*)d_in[0];
    const float* xt    = (const float*)d_in[1];
    const int*   ei_q  = (const int*)d_in[2];   // [2, E]; reference uses this for BOTH graphs
    const float* W_mlp = (const float*)d_in[4];
    const float* b_mlp = (const float*)d_in[5];
    const float* Ws    = (const float*)d_in[6]; // [L, D, D]
    const float* a_src = (const float*)d_in[7]; // [L, D]
    const float* a_dst = (const float*)d_in[8]; // [L, D]
    const float* biases= (const float*)d_in[9]; // [L, D]

    const int* esrc = ei_q;            // row 0
    const int* edst = ei_q + E_EDGES;  // row 1

    float* Xq = (float*)d_out;                        // [N, D]
    float* Xt = Xq + (size_t)N_NODES * DIM;           // [N, D]

    // workspace layout (Wpk first: 32B-aligned v16bf loads)
    __bf16*   Wpk   = (__bf16*)d_ws;                              // 4 * 16384 bf16 (128 KB)
    __bf16*   Hb    = Wpk + (size_t)NWEIGHTS * WPK_ELEMS_PER_W;   // [N, D] bf16
    float*    accum = (float*)(Hb + (size_t)N_NODES * DIM);       // [N, D] f32
    float*    as_   = accum + (size_t)N_NODES * DIM;              // [N]
    float*    ad_   = as_ + N_NODES;                              // [N]
    unsigned* m_u   = (unsigned*)(ad_ + N_NODES);                 // [N]
    float*    z     = (float*)(m_u + N_NODES);                    // [N]

    const dim3 blk(256);
    const dim3 pack_grid((NWEIGHTS * 32 * 32 + 255) / 256);          // 128 waves
    const dim3 gemm_grid(N_NODES / 16);                              // 1875 (exact)
    const dim3 node_wave_grid((N_NODES + 7) / 8);                    // 8 waves/block
    const dim3 edge_grid((EN_TOT + 255) / 256);
    const dim3 edge_wave_grid((EN_TOT + 7) / 8);
    const dim3 elem_grid((N_NODES * DIM + 255) / 256);

    // One-time (per launch) weight pre-swizzle into WMMA B layout.
    pack_weights_kernel<<<pack_grid, blk, 0, stream>>>(W_mlp, Ws, Wpk);

    // Input MLP: X = x @ W_mlp + b_mlp  (f32 + bias, straight into d_out halves)
    gemm128_wmma<false><<<gemm_grid, blk, 0, stream>>>(xq, Wpk, b_mlp, Xq, N_NODES);
    gemm128_wmma<false><<<gemm_grid, blk, 0, stream>>>(xt, Wpk, b_mlp, Xt, N_NODES);

    for (int l = 0; l < NLAYERS; ++l) {
        const __bf16* Wl = Wpk + (size_t)(l + 1) * WPK_ELEMS_PER_W;
        const float*  al = a_src  + (size_t)l * DIM;
        const float*  dl = a_dst  + (size_t)l * DIM;
        const float*  bl = biases + (size_t)l * DIM;
        float* Xs[2] = { Xq, Xt };
        for (int g = 0; g < 2; ++g) {
            float* X = Xs[g];
            gemm128_wmma<true><<<gemm_grid, blk, 0, stream>>>(X, Wl, nullptr, (void*)Hb, N_NODES);
            alpha_init_kernel<<<node_wave_grid, blk, 0, stream>>>(Hb, al, dl, as_, ad_, m_u, z, accum);
            edge_max_kernel <<<edge_grid, blk, 0, stream>>>(esrc, edst, as_, ad_, m_u);
            edge_sum_kernel <<<edge_grid, blk, 0, stream>>>(esrc, edst, as_, ad_, m_u, z);
            edge_aggr_kernel<<<edge_wave_grid, blk, 0, stream>>>(esrc, edst, as_, ad_, m_u, z, Hb, accum);
            finalize_kernel <<<elem_grid, blk, 0, stream>>>(accum, bl, X);
        }
    }
}
